// WindowAttention3D_28973849379142
// MI455X (gfx1250) — matrix operand order repaired and checked by reading, not compiled, not run on hardware
//
#include <hip/hip_runtime.h>

typedef __attribute__((ext_vector_type(16))) _Float16 v16h;
typedef __attribute__((ext_vector_type(8)))  _Float16 v8h;
typedef __attribute__((ext_vector_type(4)))  _Float16 v4h;
typedef __attribute__((ext_vector_type(8)))  float    v8f;

union V16 { v16h v; v8h h[2]; };

#define B_TOT   2048
#define NTOK    98
#define HEADS   4
#define HD      32
#define DIM     128
#define NW      64
#define ROWS    (B_TOT * NTOK)            // 200704
#define CHUNK   ((size_t)ROWS * DIM)      // 25690112 elements per output tensor
#define NPAD    112                       // 98 padded to 7 tiles of 16
#define SCALE   0.17677669529663687f      // 32^-0.5
#define QKVW_N  (384 * 128)               // 49152
#define PROJW_N (128 * 128)               // 16384

#define WMMA_F16(A, B, C) \
    __builtin_amdgcn_wmma_f32_16x16x32_f16(false, (A), false, (B), (short)0, (C), false, false)

// ---------------------------------------------------------------------------
// Stage 0: convert qkv_w / proj_w to f16 (read once, L2-resident afterwards).
// ---------------------------------------------------------------------------
__global__ __launch_bounds__(256) void cvt_w_kernel(
    const float* __restrict__ qkv_w, const float* __restrict__ proj_w,
    _Float16* __restrict__ wq, _Float16* __restrict__ wp)
{
    const int i = (blockIdx.x * 256 + threadIdx.x) * 4;
    if (i < QKVW_N) {
        float4 f = *(const float4*)&qkv_w[i];
        v4h h; h[0] = (_Float16)f.x; h[1] = (_Float16)f.y;
               h[2] = (_Float16)f.z; h[3] = (_Float16)f.w;
        *(v4h*)&wq[i] = h;
    }
    if (i < PROJW_N) {
        float4 f = *(const float4*)&proj_w[i];
        v4h h; h[0] = (_Float16)f.x; h[1] = (_Float16)f.y;
               h[2] = (_Float16)f.z; h[3] = (_Float16)f.w;
        *(v4h*)&wp[i] = h;
    }
}

// ---------------------------------------------------------------------------
// Stage 1: qkv = x @ qkv_w^T -> scatter into q (scaled), k, v (B,H,N,32).
// One block per 16-row M tile; A tile staged f16 in LDS; 8 waves x 3 N-tiles.
// ---------------------------------------------------------------------------
__global__ __launch_bounds__(256) void qkv_wmma_kernel(
    const float* __restrict__ x, const _Float16* __restrict__ wq,
    float* __restrict__ q_out, float* __restrict__ k_out, float* __restrict__ v_out)
{
    __shared__ __align__(16) _Float16 a_s[16 * 128];
    const int tid = threadIdx.x;
    const int mt  = blockIdx.x;                       // 12544 blocks

    if (tid == 0 && (mt + 1) * 16 < ROWS)
        __builtin_prefetch(&x[(size_t)(mt + 1) * 16 * DIM], 0, 1);

    for (int i = tid; i < 512; i += 256) {            // 16x128 f32 = 512 float4
        const int row = i >> 5, c4 = (i & 31) * 4;
        const int g = mt * 16 + row;
        float4 f = make_float4(0.f, 0.f, 0.f, 0.f);
        if (g < ROWS) f = *(const float4*)&x[(size_t)g * DIM + c4];
        v4h h; h[0] = (_Float16)f.x; h[1] = (_Float16)f.y;
               h[2] = (_Float16)f.z; h[3] = (_Float16)f.w;
        *(v4h*)&a_s[row * 128 + c4] = h;
    }
    __syncthreads();

    const int lane = tid & 31, wave = tid >> 5;
    const int half = lane >> 4, ln = lane & 15;
    const v8h* ar = (const v8h*)&a_s[ln * 128];

    for (int qi = 0; qi < 3; ++qi) {
        const int nt   = wave * 3 + qi;               // 24 N tiles
        const int ccol = nt * 16 + ln;
        const v8h* br  = (const v8h*)&wq[ccol * DIM];
        v8f acc = {};
        #pragma unroll
        for (int kc = 0; kc < DIM; kc += 32) {
            V16 a, b;
            a.h[0] = ar[kc / 8 + 2 * half];
            a.h[1] = ar[kc / 8 + 2 * half + 1];
            b.h[0] = br[kc / 8 + 2 * half];
            b.h[1] = br[kc / 8 + 2 * half + 1];
            acc = WMMA_F16(a.v, b.v, acc);
        }
        const int s = ccol >> 7, hh = (ccol & 127) >> 5, d = ccol & 31;
        #pragma unroll
        for (int r = 0; r < 8; ++r) {
            const int g = mt * 16 + r + 8 * half;
            if (g >= ROWS) continue;
            const int bidx = g / NTOK, n = g % NTOK;
            const size_t o = (((size_t)bidx * HEADS + hh) * NTOK + n) * HD + d;
            const float val = acc[r];
            if      (s == 0) q_out[o] = val * SCALE;
            else if (s == 1) k_out[o] = val;
            else             v_out[o] = val;
        }
    }
}

// ---------------------------------------------------------------------------
// Stage 2: per (b,h) attention. Scores/probs live as f16 in LDS; V kept
// transposed so every WMMA fragment is 2x ds_load_b128.
// ---------------------------------------------------------------------------
__global__ __launch_bounds__(256) void attn_wmma_kernel(
    const float* __restrict__ q_g, const float* __restrict__ k_g,
    const float* __restrict__ v_g, const float* __restrict__ mask,
    const float* __restrict__ rpb, const int* __restrict__ rpi,
    _Float16* __restrict__ ctx, float extra_scale)
{
    __shared__ __align__(16) _Float16 q_s[NPAD * HD];     //  7 KB
    __shared__ __align__(16) _Float16 k_s[NPAD * HD];     //  7 KB
    __shared__ __align__(16) _Float16 vT_s[HD * 128];     //  8 KB (transposed, K padded)
    __shared__ __align__(16) _Float16 p_s[NPAD * 128];    // 28 KB scores -> probs

    const int tid = threadIdx.x;
    const int b = blockIdx.x >> 2, h = blockIdx.x & 3;
    const int w = b & (NW - 1);
    const size_t base = ((size_t)b * HEADS + h) * NTOK * HD;

    for (int i = tid; i < (NPAD * HD) / 4; i += 256) {    // 896 float4
        const int row = i >> 3, c4 = (i & 7) * 4;
        float4 fq = make_float4(0.f, 0.f, 0.f, 0.f), fk = fq;
        if (row < NTOK) {
            fq = *(const float4*)&q_g[base + row * HD + c4];
            fk = *(const float4*)&k_g[base + row * HD + c4];
        }
        v4h hq, hk;
        hq[0] = (_Float16)(fq.x * extra_scale); hq[1] = (_Float16)(fq.y * extra_scale);
        hq[2] = (_Float16)(fq.z * extra_scale); hq[3] = (_Float16)(fq.w * extra_scale);
        hk[0] = (_Float16)fk.x; hk[1] = (_Float16)fk.y;
        hk[2] = (_Float16)fk.z; hk[3] = (_Float16)fk.w;
        *(v4h*)&q_s[row * HD + c4] = hq;
        *(v4h*)&k_s[row * HD + c4] = hk;
    }
    for (int i = tid; i < (128 * HD) / 4; i += 256) {     // 1024 float4 (transpose)
        const int row = i >> 3, c4 = (i & 7) * 4;
        float4 fv = make_float4(0.f, 0.f, 0.f, 0.f);
        if (row < NTOK) fv = *(const float4*)&v_g[base + row * HD + c4];
        vT_s[(c4 + 0) * 128 + row] = (_Float16)fv.x;
        vT_s[(c4 + 1) * 128 + row] = (_Float16)fv.y;
        vT_s[(c4 + 2) * 128 + row] = (_Float16)fv.z;
        vT_s[(c4 + 3) * 128 + row] = (_Float16)fv.w;
    }
    __syncthreads();

    const int lane = tid & 31, wave = tid >> 5;
    const int half = lane >> 4, ln = lane & 15;

    // ---- S = Q @ K^T : 49 tiles, single K=32 chunk, fused bias+mask ----
    for (int t = wave; t < 49; t += 8) {
        const int mt = t / 7, nt = t % 7;
        const v8h* qr = (const v8h*)&q_s[(mt * 16 + ln) * HD];
        const v8h* kr = (const v8h*)&k_s[(nt * 16 + ln) * HD];
        V16 a, bb;
        a.h[0]  = qr[2 * half];    a.h[1]  = qr[2 * half + 1];
        bb.h[0] = kr[2 * half];    bb.h[1] = kr[2 * half + 1];
        v8f c = {};
        c = WMMA_F16(a.v, bb.v, c);
        const int j = nt * 16 + ln;
        #pragma unroll
        for (int r = 0; r < 8; ++r) {
            const int i2 = mt * 16 + r + 8 * half;
            float val = -30000.f;                      // softmax padding
            if (i2 < NTOK && j < NTOK)
                val = c[r] + rpb[rpi[i2 * NTOK + j] * HEADS + h]
                           + mask[((size_t)w * NTOK + i2) * NTOK + j];
            p_s[i2 * 128 + j] = (_Float16)val;
        }
    }
    __syncthreads();

    // ---- row softmax (f32 math, f16 storage); zero padded cols ----
    if (tid < NPAD) {
        _Float16* row = &p_s[tid * 128];
        float mx = -3.0e4f;
        for (int j = 0; j < NTOK; ++j) mx = fmaxf(mx, (float)row[j]);
        float sum = 0.f;
        for (int j = 0; j < NTOK; ++j) {
            const float e = __expf((float)row[j] - mx);
            sum += e;
            row[j] = (_Float16)e;
        }
        const float inv = 1.f / sum;
        for (int j = 0; j < NTOK; ++j) row[j] = (_Float16)((float)row[j] * inv);
        for (int j = NTOK; j < 128; ++j) row[j] = (_Float16)0.f;
    }
    __syncthreads();

    // ---- ctx = P @ V : 14 tiles, K = 128 (4 chunks), all-f16 fragments ----
    for (int t = wave; t < 14; t += 8) {
        const int mt = t >> 1, nt = t & 1;
        const v8h* pr = (const v8h*)&p_s[(mt * 16 + ln) * 128];
        const v8h* vr = (const v8h*)&vT_s[(nt * 16 + ln) * 128];
        v8f c = {};
        #pragma unroll
        for (int kc = 0; kc < 128; kc += 32) {
            V16 a, bb;
            a.h[0]  = pr[kc / 8 + 2 * half];
            a.h[1]  = pr[kc / 8 + 2 * half + 1];
            bb.h[0] = vr[kc / 8 + 2 * half];
            bb.h[1] = vr[kc / 8 + 2 * half + 1];
            c = WMMA_F16(a.v, bb.v, c);
        }
        #pragma unroll
        for (int r = 0; r < 8; ++r) {
            const int i2 = mt * 16 + r + 8 * half;
            if (i2 < NTOK)
                ctx[((size_t)b * NTOK + i2) * DIM + h * HD + nt * 16 + ln] = (_Float16)c[r];
        }
    }
}

// ---------------------------------------------------------------------------
// Stage 3: out = ctx @ proj_w^T + proj_b. One block per 16-row M tile,
// 8 waves = 8 N tiles. ctx f16 lives in L2-resident workspace.
// ---------------------------------------------------------------------------
__global__ __launch_bounds__(256) void proj_wmma_kernel(
    const _Float16* __restrict__ ctx, const _Float16* __restrict__ wp,
    const float* __restrict__ proj_b, float* __restrict__ out)
{
    __shared__ __align__(16) _Float16 a_s[16 * 128];
    const int tid = threadIdx.x;
    const int mt  = blockIdx.x;                       // 12544 blocks

    {   // 16x128 halfs = 256 v8h loads, one per thread
        const int row = tid >> 4, c8 = (tid & 15) * 8;
        const int g = mt * 16 + row;
        v8h hv = {};
        if (g < ROWS) hv = *(const v8h*)&ctx[(size_t)g * DIM + c8];
        *(v8h*)&a_s[row * 128 + c8] = hv;
    }
    __syncthreads();

    const int lane = tid & 31, wave = tid >> 5;
    const int half = lane >> 4, ln = lane & 15;
    const int nt   = wave;
    const int ccol = nt * 16 + ln;
    const v8h* ar = (const v8h*)&a_s[ln * 128];
    const v8h* br = (const v8h*)&wp[ccol * DIM];

    v8f acc = {};
    #pragma unroll
    for (int kc = 0; kc < DIM; kc += 32) {
        V16 a, b;
        a.h[0] = ar[kc / 8 + 2 * half];
        a.h[1] = ar[kc / 8 + 2 * half + 1];
        b.h[0] = br[kc / 8 + 2 * half];
        b.h[1] = br[kc / 8 + 2 * half + 1];
        acc = WMMA_F16(a.v, b.v, acc);
    }
    const float bias = proj_b[ccol];
    #pragma unroll
    for (int r = 0; r < 8; ++r) {
        const int g = mt * 16 + r + 8 * half;
        if (g < ROWS) out[(size_t)g * DIM + ccol] = acc[r] + bias;
    }
}

// ---------------------------------------------------------------------------
extern "C" void kernel_launch(void* const* d_in, const int* in_sizes, int n_in,
                              void* d_out, int out_size, void* d_ws, size_t ws_size,
                              hipStream_t stream) {
    (void)in_sizes; (void)n_in; (void)out_size; (void)ws_size;
    const float* x      = (const float*)d_in[0];
    const float* mask   = (const float*)d_in[1];
    const float* prev_k = (const float*)d_in[2];
    const float* prev_v = (const float*)d_in[3];
    const float* qkv_w  = (const float*)d_in[4];
    const float* proj_w = (const float*)d_in[5];
    const float* proj_b = (const float*)d_in[6];
    const float* rpb    = (const float*)d_in[7];
    const int*   rpi    = (const int*)d_in[8];

    float* out   = (float*)d_out;
    float* x_out = out;                 // (2048, 98, 128)
    float* x2    = out + CHUNK;         // (2048, 98, 128)
    float* v_out = out + 2 * CHUNK;     // (2048, 4, 98, 32)
    float* k_out = out + 3 * CHUNK;
    float* q_out = out + 4 * CHUNK;

    _Float16* ctx = (_Float16*)d_ws;    // 25.69M f16 = 51.4 MB (reused per branch)
    _Float16* wq  = ctx + CHUNK;        // f16 qkv weights (96 KB)
    _Float16* wp  = wq + QKVW_N;        // f16 proj weights (32 KB)

    // Stage 0: f16 weight conversion (12288 threads x 4 elements)
    cvt_w_kernel<<<48, 256, 0, stream>>>(qkv_w, proj_w, wq, wp);

    // Stage 1: QKV projection, q pre-scaled.
    qkv_wmma_kernel<<<12544, 256, 0, stream>>>(x, wq, q_out, k_out, v_out);

    // Branch 1: self-attention -> x_out
    attn_wmma_kernel<<<B_TOT * HEADS, 256, 0, stream>>>(
        q_out, k_out, v_out, mask, rpb, rpi, ctx, 1.0f);
    proj_wmma_kernel<<<12544, 256, 0, stream>>>(ctx, wp, proj_b, x_out);

    // Branch 2: decoder attention (q scaled a second time) -> x2
    attn_wmma_kernel<<<B_TOT * HEADS, 256, 0, stream>>>(
        q_out, prev_k, prev_v, mask, rpb, rpi, ctx, SCALE);
    proj_wmma_kernel<<<12544, 256, 0, stream>>>(ctx, wp, proj_b, x2);
}
